// AttentionLayer_17042430231002
// MI455X (gfx1250) — matrix operand order repaired and checked by reading, not compiled, not run on hardware
//
#include <hip/hip_runtime.h>
#include <hip/hip_bf16.h>

#define DMODEL 1024
#define NHEADS 16
#define HDIM   64
#define BATCH  4
#define SEQ    1024

typedef __attribute__((ext_vector_type(16))) __bf16 v16bf;
typedef __attribute__((ext_vector_type(8)))  float  v8f;

__device__ __forceinline__ v8f wmma_bf16(v16bf a, v16bf b, v8f c) {
  // v_wmma_f32_16x16x32_bf16: (neg_a, A, neg_b, B, c_mod, C, reuse_a, reuse_b)
  return __builtin_amdgcn_wmma_f32_16x16x32_bf16(false, a, false, b, (short)0, c, false, false);
}

// ---- fragment loaders (contiguous k order: k = 16*(lane/16) + element, wave32) -------------------
// A: 16x32 tile, row-major source with leading dim ld.
// lanes 0-15: row=lane, VGPR v holds K = 2*v {,+1}
// lanes 16-31: same rows, K offset +16.
template <typename PT>
__device__ __forceinline__ v16bf load_a_frag(const PT* p, int ld) {
  const int lane = threadIdx.x & 31;
  const int row  = lane & 15;
  const int kh   = (lane >> 4) << 4;          // 0 or 16
  v16bf a;
#pragma unroll
  for (int v = 0; v < 8; ++v) {
    const int k = kh + 2 * v;
    a[2 * v]     = (__bf16)p[row * ld + k];
    a[2 * v + 1] = (__bf16)p[row * ld + k + 1];
  }
  return a;
}

// B: 32x16 tile, [k][n] row-major source with leading dim ldn.
// lanes 0-15: col=lane, K=0..15 (2 per VGPR); lanes 16-31: K=16..31.
template <typename PT>
__device__ __forceinline__ v16bf load_b_frag(const PT* p, int ldn) {
  const int lane = threadIdx.x & 31;
  const int col  = lane & 15;
  const int kh   = (lane >> 4) << 4;          // 0 or 16
  v16bf b;
#pragma unroll
  for (int v = 0; v < 8; ++v) {
    const int k = kh + 2 * v;
    b[2 * v]     = (__bf16)p[k * ldn + col];
    b[2 * v + 1] = (__bf16)p[(k + 1) * ldn + col];
  }
  return b;
}

// B^T gather: source is [n][k] row-major (e.g. torch-Linear weight [N,K]);
// contraction runs along the contiguous k axis -> 4-byte paired loads per lane.
template <typename PT>
__device__ __forceinline__ v16bf load_bt_frag(const PT* p, int ld) {
  const int lane = threadIdx.x & 31;
  const int col  = lane & 15;
  const int kh   = (lane >> 4) << 4;
  v16bf b;
#pragma unroll
  for (int v = 0; v < 8; ++v) {
    const int k = kh + 2 * v;
    b[2 * v]     = (__bf16)p[col * ld + k];
    b[2 * v + 1] = (__bf16)p[col * ld + k + 1];
  }
  return b;
}

// ---- GEMM: C[m,n] = sum_k A[m,k] * W[n,k]  (+bias) ------------------------
// Block tile 128x64, 256 threads = 8 waves, wave w -> rows w*16..w*16+15.
// A staged through LDS (fp32->bf16 convert, coalesced b128 loads);
// W fragments gathered directly from the L2-resident weight matrix.
// MODE 0: scatter bf16 into head-split [B,H,S,Dh]; MODE 1: fp32 + bias.
template <typename AT, int MODE>
__global__ __launch_bounds__(256) void gemm_wT_kernel(
    const AT* __restrict__ A, const float* __restrict__ W,
    void* __restrict__ dst, const float* __restrict__ bias) {
  __shared__ __bf16 As[128 * 32];
  const int row0 = blockIdx.y * 128;
  const int col0 = blockIdx.x * 64;
  const int tid  = threadIdx.x;
  const int wid  = tid >> 5;
  const int lane = tid & 31;

  v8f acc[4] = {v8f{}, v8f{}, v8f{}, v8f{}};

  for (int kk = 0; kk < DMODEL; kk += 32) {
    // stage A 128x32 tile as bf16
#pragma unroll
    for (int i = 0; i < 4; ++i) {
      const int le = (i << 10) + (tid << 2);
      const int r  = le >> 5;
      const int k  = le & 31;
      const AT* src = A + (size_t)(row0 + r) * DMODEL + kk + k;
      As[r * 32 + k + 0] = (__bf16)src[0];
      As[r * 32 + k + 1] = (__bf16)src[1];
      As[r * 32 + k + 2] = (__bf16)src[2];
      As[r * 32 + k + 3] = (__bf16)src[3];
    }
    if (kk + 32 < DMODEL)  // hint next A tile toward GL2
      __builtin_prefetch(A + (size_t)(row0 + (tid >> 1)) * DMODEL + kk + 32, 0, 0);
    __syncthreads();

    const v16bf a = load_a_frag(As + (wid * 16) * 32, 32);
#pragma unroll
    for (int j = 0; j < 4; ++j) {
      const v16bf b = load_bt_frag(W + (size_t)(col0 + j * 16) * DMODEL + kk, DMODEL);
      acc[j] = wmma_bf16(a, b, acc[j]);
    }
    __syncthreads();
  }

  const int colL = lane & 15;
  const int rb   = lane >> 4;
#pragma unroll
  for (int j = 0; j < 4; ++j) {
#pragma unroll
    for (int r = 0; r < 8; ++r) {
      const int m = row0 + wid * 16 + rb * 8 + r;
      const int n = col0 + j * 16 + colL;
      const float v = acc[j][r];
      if (MODE == 0) {
        const int bi = m >> 10, s = m & (SEQ - 1);
        const int h = n >> 6, d = n & (HDIM - 1);
        ((__bf16*)dst)[(((size_t)bi * NHEADS + h) * SEQ + s) * HDIM + d] = (__bf16)v;
      } else {
        ((float*)dst)[(size_t)m * DMODEL + n] = v + bias[n];
      }
    }
  }
}

// ---- scores + softmax ------------------------------------------------------
// One block per (b*h, 16-query tile). Q frags held in registers; K^T frags
// gathered from the 128KB L2-resident K panel. Each wave covers 8 key tiles of
// 16 -> full 1024-key row across the 8 waves. Softmax: shfl_xor within 16-lane
// halves + tiny LDS exchange across waves. Writes fp32 attn to d_out slice.
__global__ __launch_bounds__(256) void attn_softmax_kernel(
    const __bf16* __restrict__ qh, const __bf16* __restrict__ kh,
    const int* __restrict__ mask, float* __restrict__ attn) {
  const int bh   = blockIdx.y;          // 0..63
  const int bi   = bh >> 4;
  const int q0   = blockIdx.x * 16;
  const int tid  = threadIdx.x;
  const int wid  = tid >> 5;
  const int lane = tid & 31;
  const int colL = lane & 15;
  const int rb   = lane >> 4;

  const __bf16* Q  = qh + ((size_t)bh * SEQ + q0) * HDIM;
  const __bf16* Kp = kh + (size_t)bh * SEQ * HDIM;

  const v16bf aq0 = load_a_frag(Q, HDIM);        // k = 0..31
  const v16bf aq1 = load_a_frag(Q + 32, HDIM);   // k = 32..63

  v8f acc[8];
#pragma unroll
  for (int t = 0; t < 8; ++t) {
    const int n0 = t * 128 + wid * 16;
    const v16bf b0 = load_bt_frag(Kp + (size_t)n0 * HDIM, HDIM);
    const v16bf b1 = load_bt_frag(Kp + (size_t)n0 * HDIM + 32, HDIM);
    v8f c = {};
    c = wmma_bf16(aq0, b0, c);
    c = wmma_bf16(aq1, b1, c);
    acc[t] = c;
  }

  // scale by 1/sqrt(d_model) and apply additive mask
  const float scale = 0.03125f;  // 1/sqrt(1024)
#pragma unroll
  for (int t = 0; t < 8; ++t) {
    const int n0 = t * 128 + wid * 16;
#pragma unroll
    for (int r = 0; r < 8; ++r) {
      const int q = q0 + rb * 8 + r;
      const int kc = n0 + colL;
      float s = acc[t][r] * scale;
      s += (float)mask[((size_t)bi * SEQ + q) * SEQ + kc] * (-1e9f);
      acc[t][r] = s;
    }
  }

  __shared__ float red[16][8];

  // row max: per-lane over 8 tiles, then across the 16 lanes of this half
  float rmax[8];
#pragma unroll
  for (int r = 0; r < 8; ++r) {
    float m = -3.4e38f;
#pragma unroll
    for (int t = 0; t < 8; ++t) m = fmaxf(m, acc[t][r]);
#pragma unroll
    for (int off = 1; off < 16; off <<= 1) m = fmaxf(m, __shfl_xor(m, off, 32));
    rmax[r] = m;
  }
  if (colL == 0) {
#pragma unroll
    for (int r = 0; r < 8; ++r) red[rb * 8 + r][wid] = rmax[r];
  }
  __syncthreads();
#pragma unroll
  for (int r = 0; r < 8; ++r) {
    float m = red[rb * 8 + r][0];
#pragma unroll
    for (int j = 1; j < 8; ++j) m = fmaxf(m, red[rb * 8 + r][j]);
    rmax[r] = m;
  }
  __syncthreads();

  // exp + row sum
  float rsum[8];
#pragma unroll
  for (int r = 0; r < 8; ++r) {
    float s = 0.f;
#pragma unroll
    for (int t = 0; t < 8; ++t) {
      const float e = __expf(acc[t][r] - rmax[r]);
      acc[t][r] = e;
      s += e;
    }
#pragma unroll
    for (int off = 1; off < 16; off <<= 1) s += __shfl_xor(s, off, 32);
    rsum[r] = s;
  }
  if (colL == 0) {
#pragma unroll
    for (int r = 0; r < 8; ++r) red[rb * 8 + r][wid] = rsum[r];
  }
  __syncthreads();
  float rinv[8];
#pragma unroll
  for (int r = 0; r < 8; ++r) {
    float s = red[rb * 8 + r][0];
#pragma unroll
    for (int j = 1; j < 8; ++j) s += red[rb * 8 + r][j];
    rinv[r] = 1.0f / s;
  }

  float* out = attn + ((size_t)bh * SEQ + q0) * SEQ;
#pragma unroll
  for (int t = 0; t < 8; ++t) {
    const int n0 = t * 128 + wid * 16;
#pragma unroll
    for (int r = 0; r < 8; ++r)
      out[(size_t)(rb * 8 + r) * SEQ + n0 + colL] = acc[t][r] * rinv[r];
  }
}

// ---- context: ctx[q,d] = sum_k attn[q,k] * V[k,d]  per (b,h) ---------------
// 128 queries per block; attn fp32 staged -> bf16 LDS; V frags direct ([K][N]
// layout already matches WMMA B). Output scattered bf16 into [B,S,D].
__global__ __launch_bounds__(256) void ctx_kernel(
    const float* __restrict__ attn, const __bf16* __restrict__ vh,
    __bf16* __restrict__ ctx) {
  const int bh = blockIdx.y;
  const int bi = bh >> 4;
  const int h  = bh & 15;
  const int q0 = blockIdx.x * 128;
  __shared__ __bf16 As[128 * 32];
  const int tid = threadIdx.x, wid = tid >> 5, lane = tid & 31;

  const float*  Ap = attn + (size_t)bh * SEQ * SEQ + (size_t)q0 * SEQ;
  const __bf16* Vp = vh + (size_t)bh * SEQ * HDIM;

  v8f acc[4] = {v8f{}, v8f{}, v8f{}, v8f{}};

  for (int kk = 0; kk < SEQ; kk += 32) {
#pragma unroll
    for (int i = 0; i < 4; ++i) {
      const int le = (i << 10) + (tid << 2);
      const int r  = le >> 5;
      const int k  = le & 31;
      const float* src = Ap + (size_t)r * SEQ + kk + k;
      As[r * 32 + k + 0] = (__bf16)src[0];
      As[r * 32 + k + 1] = (__bf16)src[1];
      As[r * 32 + k + 2] = (__bf16)src[2];
      As[r * 32 + k + 3] = (__bf16)src[3];
    }
    __syncthreads();
    const v16bf a = load_a_frag(As + (wid * 16) * 32, 32);
#pragma unroll
    for (int j = 0; j < 4; ++j) {
      const v16bf b = load_b_frag(Vp + (size_t)kk * HDIM + j * 16, HDIM);
      acc[j] = wmma_bf16(a, b, acc[j]);
    }
    __syncthreads();
  }

  const int colL = lane & 15, rb = lane >> 4;
#pragma unroll
  for (int j = 0; j < 4; ++j) {
#pragma unroll
    for (int r = 0; r < 8; ++r) {
      const int q = q0 + wid * 16 + rb * 8 + r;
      const int d = j * 16 + colL;
      ctx[((size_t)bi * SEQ + q) * DMODEL + h * HDIM + d] = (__bf16)acc[j][r];
    }
  }
}

extern "C" void kernel_launch(void* const* d_in, const int* in_sizes, int n_in,
                              void* d_out, int out_size, void* d_ws, size_t ws_size,
                              hipStream_t stream) {
  const float* q    = (const float*)d_in[0];
  const float* k    = (const float*)d_in[1];
  const float* v    = (const float*)d_in[2];
  const int*   mask = (const int*)d_in[3];
  const float* Wq   = (const float*)d_in[4];
  const float* Wk   = (const float*)d_in[5];
  const float* Wv   = (const float*)d_in[6];
  const float* Wd   = (const float*)d_in[7];
  const float* bd   = (const float*)d_in[8];

  // workspace: qh | kh | vh | ctx, each B*S*D bf16 (8 MB) -> 32 MB total
  const size_t HSZ = (size_t)BATCH * NHEADS * SEQ * HDIM;  // == B*S*D
  __bf16* qh  = (__bf16*)d_ws;
  __bf16* kh  = qh + HSZ;
  __bf16* vh  = kh + HSZ;
  __bf16* ctx = vh + HSZ;

  float* out_p  = (float*)d_out;                                // [B,S,D]
  float* attn_p = out_p + (size_t)BATCH * SEQ * DMODEL;         // [B,H,S,S]

  const dim3 gGemm(DMODEL / 64, (BATCH * SEQ) / 128);           // (16, 32)
  const dim3 blk(256);

  // 1-3: head-split projections (fp32 in, bf16 head-layout out)
  gemm_wT_kernel<float, 0><<<gGemm, blk, 0, stream>>>(q, Wq, qh, nullptr);
  gemm_wT_kernel<float, 0><<<gGemm, blk, 0, stream>>>(k, Wk, kh, nullptr);
  gemm_wT_kernel<float, 0><<<gGemm, blk, 0, stream>>>(v, Wv, vh, nullptr);

  // 4: scores + softmax -> attn (fp32, second output slice)
  attn_softmax_kernel<<<dim3(SEQ / 16, BATCH * NHEADS), blk, 0, stream>>>(
      qh, kh, mask, attn_p);

  // 5: context
  ctx_kernel<<<dim3(SEQ / 128, BATCH * NHEADS), blk, 0, stream>>>(attn_p, vh, ctx);

  // 6: output projection + bias (bf16 A path)
  gemm_wT_kernel<__bf16, 1><<<gGemm, blk, 0, stream>>>(ctx, Wd, out_p, bd);
}
